// DummyModel_71734543777953
// MI455X (gfx1250) — matrix operand order repaired and checked by reading, not compiled, not run on hardware
//
#include <hip/hip_runtime.h>

#define SEQ 2048
#define HID 2048
#define NHEADS 16
#define HD 128
#define VOCAB 128
#define NLAYERS 4
#define PADTOK 127
#define ATT_SCALE 0.08838834764831845f /* 1/sqrt(128) */
#define NEG_INF (-__builtin_inff())

typedef __bf16 bf16;
typedef __attribute__((ext_vector_type(16))) __bf16 v16bf;
typedef __attribute__((ext_vector_type(8)))  __bf16 v8bf;
typedef __attribute__((ext_vector_type(8)))  float  v8f;

__device__ __forceinline__ v8f vzero8() {
  v8f z;
#pragma unroll
  for (int i = 0; i < 8; ++i) z[i] = 0.0f;
  return z;
}

// Build a 16-element bf16 fragment from two contiguous 8-element runs.
__device__ __forceinline__ v16bf load_frag2(const bf16* p0, const bf16* p1) {
  v8bf a = *(const v8bf*)p0;
  v8bf b = *(const v8bf*)p1;
  v16bf r;
#pragma unroll
  for (int i = 0; i < 8; ++i) { r[i] = a[i]; r[i + 8] = b[i]; }
  return r;
}

// ---- 16-lane row reductions via DPP row_ror rotations (VALU, no LDS) -------
__device__ __forceinline__ float rowmax16(float x) {
  x = fmaxf(x, __int_as_float(__builtin_amdgcn_update_dpp(
          0, __float_as_int(x), 0x121, 0xf, 0xf, true)));
  x = fmaxf(x, __int_as_float(__builtin_amdgcn_update_dpp(
          0, __float_as_int(x), 0x122, 0xf, 0xf, true)));
  x = fmaxf(x, __int_as_float(__builtin_amdgcn_update_dpp(
          0, __float_as_int(x), 0x124, 0xf, 0xf, true)));
  x = fmaxf(x, __int_as_float(__builtin_amdgcn_update_dpp(
          0, __float_as_int(x), 0x128, 0xf, 0xf, true)));
  return x;
}
__device__ __forceinline__ float rowsum16(float x) {
  x += __int_as_float(__builtin_amdgcn_update_dpp(
          0, __float_as_int(x), 0x121, 0xf, 0xf, true));
  x += __int_as_float(__builtin_amdgcn_update_dpp(
          0, __float_as_int(x), 0x122, 0xf, 0xf, true));
  x += __int_as_float(__builtin_amdgcn_update_dpp(
          0, __float_as_int(x), 0x124, 0xf, 0xf, true));
  x += __int_as_float(__builtin_amdgcn_update_dpp(
          0, __float_as_int(x), 0x128, 0xf, 0xf, true));
  return x;
}

// ---- async global->LDS DMA (CDNA5), tracked with ASYNCcnt ------------------
__device__ __forceinline__ void async_b128(void* lds, const bf16* g) {
  unsigned off = (unsigned)(size_t)lds;  // low 32 bits of flat LDS addr = offset
  asm volatile("global_load_async_to_lds_b128 %0, %1, off"
               :: "v"(off), "v"(g) : "memory");
}
__device__ __forceinline__ void wait_async0() {
  asm volatile("s_wait_asynccnt 0" ::: "memory");
}

// ---------------- mask auxiliaries, packed per position ---------------------
// bits [11:0]=word_int  [23:12]=seg_int  [24]=specials  [25]=seg_bound  [26]=nonpad
__global__ void aux_kernel(const int* __restrict__ tokens,
                           const int* __restrict__ segm,
                           const int* __restrict__ spec,
                           int* __restrict__ pk) {
  if (threadIdx.x == 0 && blockIdx.x == 0) {
    int csw = 0, css = 0;
    for (int t = 0; t < SEQ; ++t) {
      int spv = spec[t] != 0;
      int sbv = segm[t] == 0; // seg_bound = ~segment_mask
      int wiv = (t == 0) ? (spv + 1) : (csw + 1); // rolled cumsum+1
      int siv = (t == 0) ? (sbv + 1) : (css + 1);
      csw += spv; css += sbv;
      int npv = tokens[t] != PADTOK;
      pk[t] = wiv | (siv << 12) | (spv << 24) | (sbv << 25) | (npv << 26);
    }
  }
}

// ---------------- embedding gather (f32 -> bf16) ----------------------------
__global__ void embed_kernel(const int* __restrict__ tokens,
                             const float* __restrict__ emb_w,
                             bf16* __restrict__ dst) {
  int s = blockIdx.x;
  const float* srow = emb_w + (size_t)tokens[s] * HID;
  bf16* drow = dst + (size_t)s * HID;
  for (int i = threadIdx.x; i < HID; i += blockDim.x)
    drow[i] = (bf16)srow[i];
}

__global__ void cvt_kernel(const float* __restrict__ src,
                           bf16* __restrict__ dst, int n) {
  int i = blockIdx.x * blockDim.x + threadIdx.x;
  if (i < n) dst[i] = (bf16)src[i];
}

// ---------------- one attention layer (flash-style, bf16 WMMA) --------------
__global__ __launch_bounds__(256) void attn_kernel(
    const bf16* __restrict__ src, bf16* __restrict__ dst,
    const int* __restrict__ pk) {
  __shared__ __align__(16) bf16 Kt[2][32][136];   // [kv][d], padded stride
  __shared__ __align__(16) bf16 Vt[2][128][40];   // [d][kv], transposed
  __shared__ __align__(16) bf16 Pt[8][16][40];    // per-wave P tile [q][kv]

  const int h    = blockIdx.y;
  const int qb0  = blockIdx.x * 128;
  const int tid  = threadIdx.x;
  const int wave = tid >> 5;
  const int lane = tid & 31;
  const int ln   = lane & 15;
  const int half = lane >> 4;
  const int qb   = qb0 + wave * 16;

  // cooperative staging helpers (all 256 threads)
  auto stageK = [&](int buf, int kvb) {  // async DMA, natural [kv][d]
    int r  = tid >> 3;
    int dc = (tid & 7) * 16;
    const bf16* krow = src + (size_t)(kvb + r) * HID + h * HD + dc;
    async_b128(&Kt[buf][r][dc], krow);
    async_b128(&Kt[buf][r][dc + 8], krow + 8);
  };
  auto stageV = [&](int buf, int kvb) {  // transposed scatter [d][kv]
    int r  = tid >> 3;
    int dc = (tid & 7) * 16;
    const bf16* krow = src + (size_t)(kvb + r) * HID + h * HD + dc;
    v8bf v0 = *(const v8bf*)krow;
    v8bf v1 = *(const v8bf*)(krow + 8);
#pragma unroll
    for (int j = 0; j < 8; ++j) {
      Vt[buf][dc + j][r]     = v0[j];
      Vt[buf][dc + 8 + j][r] = v1[j];
    }
  };

  // per-row (C-layout row = r + 8*half) query-side mask data
  int q_r[8], wiq[8], siq[8], npq[8];
#pragma unroll
  for (int r = 0; r < 8; ++r) {
    int q = qb + r + half * 8;
    int p = pk[q];
    q_r[r] = q; wiq[r] = p & 0xfff; siq[r] = (p >> 12) & 0xfff;
    npq[r] = (p >> 26) & 1;
  }

  // Q fragments: A-layout, 4 chunks of K=32 over head dim
  v16bf qf[4];
  {
    const bf16* qrow = src + (size_t)(qb + ln) * HID + h * HD;
#pragma unroll
    for (int c = 0; c < 4; ++c)
      qf[c] = load_frag2(qrow + c * 32 + half * 8,
                         qrow + c * 32 + 16 + half * 8);
  }

  v8f O[8];
#pragma unroll
  for (int d = 0; d < 8; ++d) O[d] = vzero8();
  float mrow[8], lrow[8];
#pragma unroll
  for (int r = 0; r < 8; ++r) { mrow[r] = NEG_INF; lrow[r] = 0.0f; }

  const int kv_end = qb0 + 128;

  // prologue: stage tile 0
  stageK(0, 0);
  stageV(0, 0);
  wait_async0();
  __syncthreads();

  int buf = 0;
  for (int kvb = 0; kvb < kv_end; kvb += 32) {
    const int nxt = kvb + 32;
    const bool has_next = nxt < kv_end;
    if (has_next) {          // prefetch tile t+1 while computing tile t
      stageK(buf ^ 1, nxt);  // async DMA overlaps the compute below
      stageV(buf ^ 1, nxt);
    }

    if (kvb <= qb + 15) {  // wave-uniform causal skip
      // ---- S = Q K^T : two 16-wide kv tiles ----
      v8f s0 = vzero8(), s1 = vzero8();
#pragma unroll
      for (int c = 0; c < 4; ++c) {
        v16bf kb0 = load_frag2(&Kt[buf][ln][c * 32 + half * 8],
                               &Kt[buf][ln][c * 32 + 16 + half * 8]);
        v16bf kb1 = load_frag2(&Kt[buf][16 + ln][c * 32 + half * 8],
                               &Kt[buf][16 + ln][c * 32 + 16 + half * 8]);
        s0 = __builtin_amdgcn_wmma_f32_16x16x32_bf16(
                 false, qf[c], false, kb0, (short)0, s0, false, false);
        s1 = __builtin_amdgcn_wmma_f32_16x16x32_bf16(
                 false, qf[c], false, kb1, (short)0, s1, false, false);
      }

      // kv-side mask data for this lane's two columns
      const int kv0 = kvb + ln, kv1 = kvb + 16 + ln;
      const int pk0 = pk[kv0], pk1 = pk[kv1];
      const int wi0 = pk0 & 0xfff, wi1 = pk1 & 0xfff;
      const int si0 = (pk0 >> 12) & 0xfff, si1 = (pk1 >> 12) & 0xfff;
      const int sp0 = (pk0 >> 24) & 1, sp1 = (pk1 >> 24) & 1;
      const int sg0 = (pk0 >> 25) & 1, sg1 = (pk1 >> 25) & 1;
      const int np0 = (pk0 >> 26) & 1, np1 = (pk1 >> 26) & 1;

      // ---- mask + online softmax (DPP row reductions, pure VALU) ----
#pragma unroll
      for (int r = 0; r < 8; ++r) {
        bool p0 = (q_r[r] >= kv0) && npq[r] && np0 &&
                  (sp0 || wiq[r] == wi0) && (sg0 || siq[r] == si0);
        bool p1 = (q_r[r] >= kv1) && npq[r] && np1 &&
                  (sp1 || wiq[r] == wi1) && (sg1 || siq[r] == si1);
        float a = p0 ? s0[r] * ATT_SCALE : NEG_INF;
        float b = p1 ? s1[r] * ATT_SCALE : NEG_INF;
        float rm = rowmax16(fmaxf(a, b));
        float mi = fmaxf(mrow[r], rm);
        float alpha, e0, e1;
        if (mi == NEG_INF) { alpha = 1.0f; e0 = 0.0f; e1 = 0.0f; }
        else {
          alpha = __expf(mrow[r] - mi);
          e0 = p0 ? __expf(a - mi) : 0.0f;
          e1 = p1 ? __expf(b - mi) : 0.0f;
        }
        float rs = rowsum16(e0 + e1);
        lrow[r] = lrow[r] * alpha + rs;
        mrow[r] = mi;
#pragma unroll
        for (int d = 0; d < 8; ++d) O[d][r] *= alpha;
        Pt[wave][r + half * 8][ln]      = (bf16)e0;
        Pt[wave][r + half * 8][ln + 16] = (bf16)e1;
      }

      // ---- O += P V (P round-trips LDS: C-layout -> A-layout) ----
      v16bf pf = load_frag2(&Pt[wave][ln][half * 8],
                            &Pt[wave][ln][16 + half * 8]);
#pragma unroll
      for (int d = 0; d < 8; ++d) {
        v16bf vb = load_frag2(&Vt[buf][d * 16 + ln][half * 8],
                              &Vt[buf][d * 16 + ln][16 + half * 8]);
        O[d] = __builtin_amdgcn_wmma_f32_16x16x32_bf16(
                   false, pf, false, vb, (short)0, O[d], false, false);
      }
    }

    wait_async0();     // prefetched K tile landed in LDS
    __syncthreads();   // all waves done reading buf, staging of buf^1 visible
    buf ^= 1;
  }

  // ---- epilogue: normalize, write bf16 for next layer ----
#pragma unroll
  for (int r = 0; r < 8; ++r) {
    float inv = (lrow[r] > 0.0f) ? (1.0f / lrow[r]) : 0.0f;
    bf16* orow = dst + (size_t)q_r[r] * HID + h * HD;
#pragma unroll
    for (int d = 0; d < 8; ++d)
      orow[d * 16 + ln] = (bf16)(O[d][r] * inv);
  }
}

// ---------------- final vocab GEMM: logits = hid @ lm_w^T + b ---------------
__global__ __launch_bounds__(256) void logits_kernel(
    const bf16* __restrict__ hid, const bf16* __restrict__ w,
    const float* __restrict__ bias, float* __restrict__ out) {
  const int tid  = threadIdx.x;
  const int wave = tid >> 5;
  const int lane = tid & 31;
  const int ln   = lane & 15;
  const int half = lane >> 4;
  const int mb = blockIdx.x * 128 + wave * 16;
  const int nb = blockIdx.y * 16;

  v8f acc = vzero8();
  const bf16* arow = hid + (size_t)(mb + ln) * HID;
  const bf16* brow = w + (size_t)(nb + ln) * HID;  // lm_w is [vocab][HID] = B N-major
  for (int c = 0; c < HID / 32; ++c) {
    v16bf af = load_frag2(arow + c * 32 + half * 8,
                          arow + c * 32 + 16 + half * 8);
    v16bf bf_ = load_frag2(brow + c * 32 + half * 8,
                           brow + c * 32 + 16 + half * 8);
    acc = __builtin_amdgcn_wmma_f32_16x16x32_bf16(
              false, af, false, bf_, (short)0, acc, false, false);
  }
  float bcol = bias[nb + ln];
#pragma unroll
  for (int r = 0; r < 8; ++r)
    out[(size_t)(mb + r + half * 8) * VOCAB + nb + ln] = acc[r] + bcol;
}

// ---------------- tiny segment head ----------------------------------------
__global__ void seg_kernel(const bf16* __restrict__ hid,
                           const float* __restrict__ seg_w,
                           const float* __restrict__ seg_b,
                           float* __restrict__ out) {
  int i = blockIdx.x * blockDim.x + threadIdx.x; // 0..4095
  if (i >= SEQ * 2) return;
  int s = i >> 1, j = i & 1;
  const bf16* hrow = hid + (size_t)s * HID;
  const float* wrow = seg_w + (size_t)j * HID;
  float acc = 0.0f;
  for (int k = 0; k < HID; ++k) acc += (float)hrow[k] * wrow[k];
  out[(size_t)SEQ * VOCAB + i] = acc + seg_b[j];
}

extern "C" void kernel_launch(void* const* d_in, const int* in_sizes, int n_in,
                              void* d_out, int out_size, void* d_ws, size_t ws_size,
                              hipStream_t stream) {
  const int*   tokens = (const int*)d_in[0];
  const int*   segm   = (const int*)d_in[1];
  const int*   spec   = (const int*)d_in[2];
  const float* emb_w  = (const float*)d_in[3];
  const float* lm_w   = (const float*)d_in[4];
  const float* lm_b   = (const float*)d_in[5];
  const float* seg_w  = (const float*)d_in[6];
  const float* seg_b  = (const float*)d_in[7];
  float* out = (float*)d_out;

  char* ws = (char*)d_ws;
  int*  pk  = (int*)(ws + 0);                       // 8 KB packed mask aux
  const size_t HBYTES = (size_t)SEQ * HID * sizeof(bf16); // 8 MB
  bf16* hA  = (bf16*)(ws + 40960);
  bf16* hB  = (bf16*)(ws + 40960 + HBYTES);
  bf16* wbf = (bf16*)(ws + 40960 + 2 * HBYTES);

  aux_kernel<<<1, 32, 0, stream>>>(tokens, segm, spec, pk);
  embed_kernel<<<SEQ, 256, 0, stream>>>(tokens, emb_w, hA);
  cvt_kernel<<<(VOCAB * HID + 255) / 256, 256, 0, stream>>>(lm_w, wbf, VOCAB * HID);

  bf16* src = hA; bf16* dst = hB;
  dim3 agrid(SEQ / 128, NHEADS);
  for (int l = 0; l < NLAYERS; ++l) {
    attn_kernel<<<agrid, 256, 0, stream>>>(src, dst, pk);
    bf16* t = src; src = dst; dst = t;
  }
  logits_kernel<<<dim3(SEQ / 128, VOCAB / 16), 256, 0, stream>>>(src, wbf, lm_b, out);
  seg_kernel<<<(SEQ * 2 + 255) / 256, 256, 0, stream>>>(src, seg_w, seg_b, out);
}
